// MaskedAttentionHead_30837865185846
// MI455X (gfx1250) — matrix operand order repaired and checked by reading, not compiled, not run on hardware
//
#include <hip/hip_runtime.h>
#include <stdint.h>

// Sizes fixed by the reference problem.
#define BB 4
#define TT 4096
#define CC 1024
#define HH 64
#define SW 64          // key/value tile width in attention

typedef __attribute__((ext_vector_type(16))) __bf16 v16bf;
typedef __attribute__((ext_vector_type(8)))  float  v8f;

union FragB {
    v16bf    v;
    __bf16   b[16];
    uint16_t h[16];
    uint4    q[2];
};

// native f32 -> bf16 (RNE via hardware cvt)
__device__ __forceinline__ uint16_t f2bf(float f) {
    union { __bf16 h; uint16_t u; } c;
    c.h = (__bf16)f;
    return c.u;
}

// ---- async global->LDS staging with sync fallback --------------------------
#if __has_builtin(__builtin_amdgcn_global_load_async_to_lds_b128)
#define ASYNC_CP 1
#else
#define ASYNC_CP 0
#endif

#if ASYNC_CP
typedef int vint4 __attribute__((vector_size(16)));
typedef __attribute__((address_space(1))) vint4* as1_int4p;
typedef __attribute__((address_space(3))) vint4* as3_int4p;
#endif

__device__ __forceinline__ void cp16(uint16_t* dst_lds, const uint16_t* src_glb) {
#if ASYNC_CP
    __builtin_amdgcn_global_load_async_to_lds_b128(
        (as1_int4p)src_glb, (as3_int4p)dst_lds, 0, 0);
#else
    *(uint4*)dst_lds = *(const uint4*)src_glb;
#endif
}

__device__ __forceinline__ void cp_wait() {
#if ASYNC_CP
#if __has_builtin(__builtin_amdgcn_s_wait_asynccnt)
    __builtin_amdgcn_s_wait_asynccnt(0);
#else
    asm volatile("s_wait_asynccnt 0" ::: "memory");
#endif
#endif
}

// ---------------------------------------------------------------------------
// Kernel 0: transpose + downconvert weights: wT[w][h][c] = bf16(W_w[c][h])
// w: 0=q, 1=k, 2=v
// ---------------------------------------------------------------------------
__global__ void wt_kernel(const float* __restrict__ Wq,
                          const float* __restrict__ Wk,
                          const float* __restrict__ Wv,
                          uint16_t* __restrict__ wT) {
    int idx = blockIdx.x * blockDim.x + threadIdx.x;  // 3*H*C threads exactly
    int w = idx / (HH * CC);
    int h = (idx / CC) % HH;
    int c = idx % CC;
    const float* W = (w == 0) ? Wq : (w == 1) ? Wk : Wv;
    wT[idx] = f2bf(W[c * HH + h]);
}

// ---------------------------------------------------------------------------
// Kernel 1: fused q/k/v projection with WMMA bf16.
// One wave computes a 16-row tile of all three projections.
// q stored pre-scaled by (1/sqrt(H))*log2(e) so attention scores come out of
// the WMMA already in the exp2 domain.
// q,k stored [B,T,H] bf16 row-major; v stored transposed [B,H,T] bf16.
// ---------------------------------------------------------------------------
__global__ __launch_bounds__(128) void proj_kernel(
    const float*    __restrict__ x,
    const uint16_t* __restrict__ wT,
    const float*    __restrict__ bq,
    const float*    __restrict__ bk,
    const float*    __restrict__ bv,
    uint16_t* __restrict__ qO,
    uint16_t* __restrict__ kO,
    uint16_t* __restrict__ vTO) {

    const int lane = threadIdx.x & 31;
    const int wave = threadIdx.x >> 5;
    const int gw   = blockIdx.x * 4 + wave;          // 0 .. B*T/16-1
    const int b    = gw / (TT / 16);
    const int trow = (gw % (TT / 16)) * 16;
    const int hi   = lane >> 4;                      // half-wave select
    const int ln   = lane & 15;

    const v8f vzero = {0.f, 0.f, 0.f, 0.f, 0.f, 0.f, 0.f, 0.f};
    v8f acc[3][4];
#pragma unroll
    for (int w = 0; w < 3; ++w)
#pragma unroll
        for (int nt = 0; nt < 4; ++nt) acc[w][nt] = vzero;

    const int t = trow + ln;                         // A-matrix row for this lane
    const float* xrow = x + (size_t)(b * TT + t) * CC;

    for (int c0 = 0; c0 < CC; c0 += 32) {
        // ---- A fragment: 16x32 bf16 tile of x (fp32 -> bf16 in regs) ----
        // lanes 0-15: K = c0+{0..7,16..23}; lanes 16-31: K = c0+{8..15,24..31}
        FragB a;
        const int cb = c0 + (hi ? 8 : 0);
        float4 f0 = *(const float4*)(xrow + cb);
        float4 f1 = *(const float4*)(xrow + cb + 4);
        float4 f2 = *(const float4*)(xrow + cb + 16);
        float4 f3 = *(const float4*)(xrow + cb + 20);
        a.b[0]  = (__bf16)f0.x; a.b[1]  = (__bf16)f0.y; a.b[2]  = (__bf16)f0.z; a.b[3]  = (__bf16)f0.w;
        a.b[4]  = (__bf16)f1.x; a.b[5]  = (__bf16)f1.y; a.b[6]  = (__bf16)f1.z; a.b[7]  = (__bf16)f1.w;
        a.b[8]  = (__bf16)f2.x; a.b[9]  = (__bf16)f2.y; a.b[10] = (__bf16)f2.z; a.b[11] = (__bf16)f2.w;
        a.b[12] = (__bf16)f3.x; a.b[13] = (__bf16)f3.y; a.b[14] = (__bf16)f3.z; a.b[15] = (__bf16)f3.w;

        // ---- B fragments grouped 4 at a time so loads overlap WMMAs ----
#pragma unroll
        for (int w = 0; w < 3; ++w) {
            FragB bb[4];
#pragma unroll
            for (int nt = 0; nt < 4; ++nt) {
                const uint16_t* bp =
                    wT + ((size_t)w * HH + nt * 16 + ln) * CC + c0 + (hi ? 8 : 0);
                bb[nt].q[0] = *(const uint4*)bp;
                bb[nt].q[1] = *(const uint4*)(bp + 16);
            }
#pragma unroll
            for (int nt = 0; nt < 4; ++nt)
                acc[w][nt] = __builtin_amdgcn_wmma_f32_16x16x32_bf16(
                    false, a.v, false, bb[nt].v, (short)0, acc[w][nt], false, false);
        }
    }

    // ---- bias + store (q pre-scaled; q,k row major; v transposed) ----
    const float QSCALE = 0.18033688011112042f;   // (1/sqrt(64)) * log2(e)
    const float* biases[3] = {bq, bk, bv};
#pragma unroll
    for (int w = 0; w < 3; ++w) {
#pragma unroll
        for (int nt = 0; nt < 4; ++nt) {
            const int   n    = nt * 16 + ln;
            const float bias = biases[w][n];
#pragma unroll
            for (int i = 0; i < 8; ++i) {
                const int tm = trow + i + 8 * hi;
                float val = acc[w][nt][i] + bias;
                if (w == 0) {
                    qO[(size_t)(b * TT + tm) * HH + n] = f2bf(val * QSCALE);
                } else if (w == 1) {
                    kO[(size_t)(b * TT + tm) * HH + n] = f2bf(val);
                } else {
                    vTO[(size_t)(b * HH + n) * TT + tm] = f2bf(val);
                }
            }
        }
    }
}

// ---------------------------------------------------------------------------
// Kernel 2: causal flash attention. Block = 4 waves; wave w owns query rows
// q0..q0+15. 64-wide K/V tiles staged in LDS (async when available), online
// softmax in registers (denominator accumulated by WMMA against a ones
// matrix as a 5th output tile), mask-free fast path off the diagonal.
// ---------------------------------------------------------------------------
__global__ __launch_bounds__(128) void attn_kernel(
    const uint16_t* __restrict__ qI,
    const uint16_t* __restrict__ kI,
    const uint16_t* __restrict__ vTI,
    float* __restrict__ out) {

    __shared__ uint16_t kS[64 * 72];      // [s][h], row stride 72 halfs (144 B)
    __shared__ uint16_t vS[64 * 72];      // [h][s], row stride 72 halfs
    __shared__ uint16_t pS[4][16 * 72];   // per-wave P bounce, [m][n] stride 72

    const int lane = threadIdx.x & 31;
    const int wave = threadIdx.x >> 5;
    const int b    = blockIdx.x / (TT / 64);
    const int blk  = blockIdx.x % (TT / 64);
    const int q0   = blk * 64 + wave * 16;
    const int hi   = lane >> 4;
    const int ln   = lane & 15;
    const int sr   = threadIdx.x >> 1;          // staging row 0..63
    const int sch  = (threadIdx.x & 1) * 32;    // staging chunk (halfs)

    // ---- Q fragments: 2 k-steps over H=64, loaded once (pre-scaled) ----
    FragB qf[2];
    {
        const uint16_t* qrow = qI + (size_t)(b * TT + q0 + ln) * HH;
#pragma unroll
        for (int j = 0; j < 2; ++j) {
            const int h0 = j * 32 + (hi ? 8 : 0);
            qf[j].q[0] = *(const uint4*)(qrow + h0);
            qf[j].q[1] = *(const uint4*)(qrow + h0 + 16);
        }
    }

    // ones B-fragment for the denominator WMMA (bf16 1.0 = 0x3F80)
    FragB ones;
#pragma unroll
    for (int z = 0; z < 16; ++z) ones.h[z] = 0x3F80u;

    const v8f vzero = {0.f, 0.f, 0.f, 0.f, 0.f, 0.f, 0.f, 0.f};
    v8f o[5];                       // o[0..3] = output tiles, o[4] = row sums
#pragma unroll
    for (int nt = 0; nt < 5; ++nt) o[nt] = vzero;
    float ml[8];
#pragma unroll
    for (int i = 0; i < 8; ++i) ml[i] = -INFINITY;

    const int send = blk * 64 + 64;             // causal upper bound for block

    for (int s0 = 0; s0 < send; s0 += SW) {
        __syncthreads();
        // ---- stage K[64][64] and Vt[64][64] bf16 tiles into LDS ----
        {
            const uint16_t* ks = kI + (size_t)(b * TT + s0 + sr) * HH + sch;
            uint16_t*       kd = kS + sr * 72 + sch;
            const uint16_t* vs = vTI + (size_t)(b * HH + sr) * TT + s0 + sch;
            uint16_t*       vd = vS + sr * 72 + sch;
#pragma unroll
            for (int c8 = 0; c8 < 32; c8 += 8) {
                cp16(kd + c8, ks + c8);
                cp16(vd + c8, vs + c8);
            }
            cp_wait();
        }
        __syncthreads();

        // prefetch next tile while we compute on this one
        if (s0 + SW < send) {
            __builtin_prefetch(kI + (size_t)(b * TT + s0 + SW + sr) * HH + sch, 0, 0);
            __builtin_prefetch(vTI + (size_t)(b * HH + sr) * TT + s0 + SW + sch, 0, 0);
        }

        // ---- scores: Q(16x64) @ K^T(64x64) -> four 16x16 C frags ----
        v8f c[4];
#pragma unroll
        for (int nt = 0; nt < 4; ++nt) c[nt] = vzero;
#pragma unroll
        for (int j = 0; j < 2; ++j) {
            FragB kb[4];
#pragma unroll
            for (int nt = 0; nt < 4; ++nt) {
                const uint16_t* kp = kS + (nt * 16 + ln) * 72 + j * 32 + (hi ? 8 : 0);
                kb[nt].q[0] = *(const uint4*)kp;
                kb[nt].q[1] = *(const uint4*)(kp + 16);
            }
#pragma unroll
            for (int nt = 0; nt < 4; ++nt)
                c[nt] = __builtin_amdgcn_wmma_f32_16x16x32_bf16(
                    false, qf[j].v, false, kb[nt].v, (short)0, c[nt], false, false);
        }

        // masking only needed on the diagonal tile for this wave's rows
        const bool diag = (s0 + SW - 1 > q0);

        // ---- online softmax (scores already in exp2 domain) ----
#pragma unroll
        for (int i = 0; i < 8; ++i) {
            const int trow = q0 + i + 8 * hi;
            float x0 = c[0][i], x1 = c[1][i], x2 = c[2][i], x3 = c[3][i];
            if (diag) {
                if (s0 + ln      > trow) x0 = -INFINITY;
                if (s0 + ln + 16 > trow) x1 = -INFINITY;
                if (s0 + ln + 32 > trow) x2 = -INFINITY;
                if (s0 + ln + 48 > trow) x3 = -INFINITY;
            }
            float rm = fmaxf(fmaxf(x0, x1), fmaxf(x2, x3));
#pragma unroll
            for (int d = 1; d < 16; d <<= 1)
                rm = fmaxf(rm, __shfl_xor(rm, d, 32));   // stays in 16-lane group
            const float mnew = fmaxf(ml[i], rm);
            const float al = __builtin_amdgcn_exp2f(ml[i] - mnew);
            ml[i] = mnew;
            const float p0 = __builtin_amdgcn_exp2f(x0 - mnew);
            const float p1 = __builtin_amdgcn_exp2f(x1 - mnew);
            const float p2 = __builtin_amdgcn_exp2f(x2 - mnew);
            const float p3 = __builtin_amdgcn_exp2f(x3 - mnew);
#pragma unroll
            for (int nt = 0; nt < 5; ++nt) o[nt][i] *= al;   // incl. row sums
            // stage P row-major bf16 for C->A layout transpose
            uint16_t* pr = &pS[wave][(i + 8 * hi) * 72 + ln];
            pr[0]  = f2bf(p0);
            pr[16] = f2bf(p1);
            pr[32] = f2bf(p2);
            pr[48] = f2bf(p3);
        }
        __builtin_amdgcn_wave_barrier();
        asm volatile("s_wait_dscnt 0" ::: "memory");   // wave-local LDS RAW

        // ---- O += P(16x64) @ V(64x64), rowsum += P @ ones: 10 WMMAs ----
        FragB pa[2];
#pragma unroll
        for (int j = 0; j < 2; ++j) {
            const uint16_t* pp = &pS[wave][ln * 72 + j * 32 + (hi ? 8 : 0)];
            pa[j].q[0] = *(const uint4*)pp;
            pa[j].q[1] = *(const uint4*)(pp + 16);
        }
#pragma unroll
        for (int j = 0; j < 2; ++j) {
            FragB vb[4];
#pragma unroll
            for (int nt = 0; nt < 4; ++nt) {
                const uint16_t* vp = vS + (nt * 16 + ln) * 72 + j * 32 + (hi ? 8 : 0);
                vb[nt].q[0] = *(const uint4*)vp;
                vb[nt].q[1] = *(const uint4*)(vp + 16);
            }
#pragma unroll
            for (int nt = 0; nt < 4; ++nt)
                o[nt] = __builtin_amdgcn_wmma_f32_16x16x32_bf16(
                    false, pa[j].v, false, vb[nt].v, (short)0, o[nt], false, false);
            o[4] = __builtin_amdgcn_wmma_f32_16x16x32_bf16(
                false, pa[j].v, false, ones.v, (short)0, o[4], false, false);
        }
    }

    // ---- epilogue: out = O / rowsum, fp32 ----
#pragma unroll
    for (int i = 0; i < 8; ++i) {
        const float inv = 1.0f / o[4][i];
        const int   t   = q0 + i + 8 * hi;
        float* orow = out + (size_t)(b * TT + t) * HH;
#pragma unroll
        for (int nt = 0; nt < 4; ++nt)
            orow[nt * 16 + ln] = o[nt][i] * inv;
    }
}

// ---------------------------------------------------------------------------
extern "C" void kernel_launch(void* const* d_in, const int* in_sizes, int n_in,
                              void* d_out, int out_size, void* d_ws, size_t ws_size,
                              hipStream_t stream) {
    (void)in_sizes; (void)n_in; (void)out_size; (void)ws_size;
    const float* x  = (const float*)d_in[0];
    const float* Wk = (const float*)d_in[1];
    const float* bk = (const float*)d_in[2];
    const float* Wq = (const float*)d_in[3];
    const float* bq = (const float*)d_in[4];
    const float* Wv = (const float*)d_in[5];
    const float* bv = (const float*)d_in[6];
    float* out = (float*)d_out;

    // workspace layout (bytes)
    char* ws = (char*)d_ws;
    uint16_t* wT = (uint16_t*)(ws);                                  // 3*H*C bf16
    uint16_t* qW = (uint16_t*)(ws + 393216);                         // B*T*H bf16
    uint16_t* kW = (uint16_t*)(ws + 393216 + 2097152);               // B*T*H bf16
    uint16_t* vW = (uint16_t*)(ws + 393216 + 2 * 2097152);           // B*H*T bf16

    wt_kernel<<<dim3(3 * HH * CC / 256), dim3(256), 0, stream>>>(Wq, Wk, Wv, wT);
    proj_kernel<<<dim3(BB * TT / 16 / 4), dim3(128), 0, stream>>>(
        x, wT, bq, bk, bv, qW, kW, vW);
    attn_kernel<<<dim3(BB * TT / 64), dim3(128), 0, stream>>>(qW, kW, vW, out);
}
